// GRU_Attention_Sentence_66073776882265
// MI455X (gfx1250) — compile-verified, matrix operand order checked by reading, not run on hardware
//
#include <hip/hip_runtime.h>
#include <hip/hip_bf16.h>
#include <math.h>

// ---------------- problem constants ----------------
#define BB    64      // batch
#define SS    128     // seq len
#define VEC   300
#define VECP  320     // VEC padded to mult of 32 (K for input GEMM)
#define HID   256
#define G3    768     // 3*HID
#define H2    512     // 2*HID
#define ATT   500
#define ATTP  512     // ATT padded to mult of 16 rows
#define DOM   16

typedef __attribute__((ext_vector_type(16))) _Float16 v16h;
typedef __attribute__((ext_vector_type(8)))  float    v8f;
typedef unsigned int u32x4 __attribute__((ext_vector_type(4)));
typedef int          i32x8 __attribute__((ext_vector_type(8)));
typedef int          i32x4 __attribute__((ext_vector_type(4)));

#define WMMA_F16(a, b, c) \
  __builtin_amdgcn_wmma_f32_16x16x32_f16(false, (a), false, (b), (short)0, (c), false, false)

// Load a 16x32 f16 fragment (A, or B given as rows of W == columns of B)
// from row-major memory with leading dimension ld (elements).
// Per ISA 7.12.2: lane l (m = l&15, kh = (l>>4)*8):
//   elems 0..7  = row[m][kh + 0..7]
//   elems 8..15 = row[m][16 + kh + 0..7]
__device__ __forceinline__ v16h load_frag(const _Float16* base, int ld) {
  int lane = threadIdx.x & 31;
  const _Float16* row = base + (size_t)(lane & 15) * ld + ((lane >> 4) << 3);
  v16h f;
  reinterpret_cast<float4*>(&f)[0] = *reinterpret_cast<const float4*>(row);
  reinterpret_cast<float4*>(&f)[1] = *reinterpret_cast<const float4*>(row + 16);
  return f;
}

// Store 16x16 f32 D tile (+ per-column bias) to row-major memory.
// Lane l holds column n = l&15, rows (l>>4)*8 + r for r in 0..7 (VGPR r).
__device__ __forceinline__ void store_tile_bias(float* p, int ld, v8f d, const float* bias) {
  int lane = threadIdx.x & 31;
  int n  = lane & 15;
  int rb = (lane >> 4) * 8;
  float bv = bias[n];
#pragma unroll
  for (int r = 0; r < 8; ++r) p[(size_t)(rb + r) * ld + n] = d[r] + bv;
}

__device__ __forceinline__ float sigmf(float x) { return 1.f / (1.f + __expf(-x)); }

// TDM: stage a [rows x 320] f16 slab (row-major, stride 320) from global into
// contiguous LDS.  D# per CDNA5 ISA ch.8: group0 {count=1, lds_addr, global_addr,
// type=2}; group1 {data_size=2B, tensor_dim0=320, tensor_dim1=rows,
// tile_dim0=320, tile_dim1=rows, dim0_stride=320}.  Groups 2/3 unused (2D).
// clang-23 lane: 6-arg builtin (g0, g1, g2, g3, extra i32x8, cpol).
__device__ __forceinline__ void tdm_load_slab(_Float16* lds_dst, const _Float16* gsrc, int rows) {
  unsigned long long ga = (unsigned long long)(const void*)gsrc;
  unsigned int lo = (unsigned int)(unsigned long long)(void*)lds_dst;  // LDS byte offset (addr[31:0])
  u32x4 g0;
  g0[0] = 1u;                                        // count=1, no gather
  g0[1] = lo;                                        // lds_addr
  g0[2] = (unsigned int)ga;                          // global_addr[31:0]
  g0[3] = (unsigned int)((ga >> 32) & 0x1FFFFFFu) | (2u << 30);  // addr[56:32] | type=2
  i32x8 g1;
  g1[0] = (int)(1u << 16);                           // data_size=1 (2 bytes)
  g1[1] = (int)((VECP & 0xFFFF) << 16);              // tensor_dim0[15:0] @ bits63:48
  g1[2] = (int)(((VECP >> 16) & 0xFFFF) | ((rows & 0xFFFF) << 16)); // dim0 hi | tensor_dim1 lo
  g1[3] = (int)(((rows >> 16) & 0xFFFF) | ((VECP & 0xFFFF) << 16)); // dim1 hi | tile_dim0
  g1[4] = (int)(rows & 0xFFFF);                      // tile_dim1 | tile_dim2=0
  g1[5] = (int)VECP;                                 // tensor_dim0_stride[31:0]
  g1[6] = 0;                                         // stride hi | dim1_stride lo
  g1[7] = 0;
  i32x4 gz4 = {0, 0, 0, 0};
  i32x8 gz8 = {0, 0, 0, 0, 0, 0, 0, 0};
  __builtin_amdgcn_tensor_load_to_lds(g0, g1, gz4, gz4, gz8, 0);
  __builtin_amdgcn_s_wait_tensorcnt(0);
}

// ---------------- K1: embedding gather -> xs_h [S][B][VECP] f16 ----------------
__global__ void embed_kernel(const int* __restrict__ x, const float* __restrict__ emb,
                             _Float16* __restrict__ xs_h) {
  int i = blockIdx.x * blockDim.x + threadIdx.x;
  const int total = SS * BB * VECP;
  if (i >= total) return;
  int k = i % VECP;
  int row = i / VECP;          // row = s*B + b
  int b = row % BB;
  int s = row / BB;
  float v = 0.f;
  if (k < VEC) {
    int tok = x[b * SS + s];   // x is [B,S]
    v = emb[(size_t)tok * VEC + k];
  }
  xs_h[i] = (_Float16)v;
}

// ---------------- K2: generic fp32 -> f16 with zero padding ----------------
__global__ void pad_convert(const float* __restrict__ src, _Float16* __restrict__ dst,
                            int srows, int scols, int drows, int dcols) {
  int i = blockIdx.x * blockDim.x + threadIdx.x;
  int total = drows * dcols;
  if (i >= total) return;
  int r = i / dcols, c = i % dcols;
  float v = (r < srows && c < scols) ? src[(size_t)r * scols + c] : 0.f;
  dst[i] = (_Float16)v;
}

// ---------------- K3: input projection GEMM  gi[dir][s][b][g] ----------------
// M = S*B = 8192, N = 768, K = VECP = 320.  Wave tile: 16(M) x 64(N).
// The 64x320 f16 W slab shared by the whole block is staged into LDS by the
// Tensor Data Mover once; B fragments then come from LDS.
// grid = (12, 64, 2), block = 256 (8 waves).
__global__ void __launch_bounds__(256) igemm_kernel(
    const _Float16* __restrict__ xs_h, const _Float16* __restrict__ Wih_h,
    const float* __restrict__ bih_f, const float* __restrict__ bih_b,
    float* __restrict__ gi) {
  __shared__ _Float16 wslab[64 * VECP];              // 40 KB
  int wave = threadIdx.x >> 5;
  int dir  = blockIdx.z;
  int n0   = blockIdx.x * 64;
  int mtile = blockIdx.y * 8 + wave;            // 0..511
  int m0 = mtile * 16;
  int s  = m0 >> 6;                             // 4 tiles per s (B=64)
  int b0 = m0 & 63;
  int ssrc = dir ? (SS - 1 - s) : s;            // backward dir uses reversed xs
  const _Float16* A = xs_h + ((size_t)ssrc * BB + b0) * VECP;
  const _Float16* W = Wih_h + (size_t)dir * G3 * VECP + (size_t)n0 * VECP;
  const float* bih = dir ? bih_b : bih_f;

  if (threadIdx.x < 32) {                       // wave 0 issues the TDM copy
    tdm_load_slab(wslab, W, 64);
  }
  __syncthreads();

  v8f acc[4] = {};
  for (int kb = 0; kb < VECP / 32; ++kb) {
    v16h a = load_frag(A + kb * 32, VECP);
#pragma unroll
    for (int t = 0; t < 4; ++t) {
      v16h bf = load_frag(wslab + (size_t)(t * 16) * VECP + kb * 32, VECP);
      acc[t] = WMMA_F16(a, bf, acc[t]);
    }
  }
  float* out = gi + (size_t)dir * SS * BB * G3 + ((size_t)s * BB + b0) * G3 + n0;
#pragma unroll
  for (int t = 0; t < 4; ++t) store_tile_bias(out + t * 16, G3, acc[t], bih + n0 + t * 16);
}

// ---------------- K4: GRU scan (sequential over S) ----------------
// grid = 8 (dir = bid>>2, mblk = bid&3: 16 batch rows each), block = 512 (16 waves).
// h state (16 x 256) double-buffered in LDS as f16; per step each wave computes
// the r/z/n 16x16 tiles for one hidden-column block and applies gates in regs.
__global__ void __launch_bounds__(512) gru_scan_kernel(
    const float* __restrict__ gi, const _Float16* __restrict__ Whh_h,
    const float* __restrict__ bhh_f, const float* __restrict__ bhh_b,
    float* __restrict__ hcat, _Float16* __restrict__ hcat_h) {
  __shared__ _Float16 h16[2][16][264];          // 264 = 256 + 8 pad (16B-aligned rows)
  int tid = threadIdx.x;
  int lane = tid & 31, wave = tid >> 5;         // wave 0..15
  int dir  = blockIdx.x >> 2;
  int mblk = blockIdx.x & 3;

  for (int i = tid; i < 2 * 16 * 264; i += 512) (&h16[0][0][0])[i] = (_Float16)0.f;
  __syncthreads();

  const float*    giD = gi + (size_t)dir * SS * BB * G3;
  const _Float16* Whh = Whh_h + (size_t)dir * G3 * HID;
  const float*    bhh = dir ? bhh_b : bhh_f;

  int ncol = lane & 15;
  int rb   = (lane >> 4) * 8;
  int j0   = wave * 16;                         // this wave's hidden-column block
  int j    = j0 + ncol;
  float br = bhh[j], bz = bhh[256 + j], bn = bhh[512 + j];

  for (int s = 0; s < SS; ++s) {
    int cur = s & 1, nxt = cur ^ 1;
    int sout = dir ? (SS - 1 - s) : s;

    // prefetch next step's gate inputs while this step's GEMM runs
    if (s + 1 < SS) {
      const float* nx = giD + ((size_t)(s + 1) * BB + (mblk * 16 + ncol)) * G3 + j0;
      __builtin_prefetch(nx, 0, 0);
      __builtin_prefetch(nx + 256, 0, 0);
      __builtin_prefetch(nx + 512, 0, 0);
    }

    v8f cr = {}, cz = {}, cn = {};
    for (int kb = 0; kb < HID / 32; ++kb) {
      v16h a   = load_frag(&h16[cur][0][0] + kb * 32, 264);
      v16h br_ = load_frag(Whh + (size_t)(j0)       * HID + kb * 32, HID);
      v16h bz_ = load_frag(Whh + (size_t)(256 + j0) * HID + kb * 32, HID);
      v16h bn_ = load_frag(Whh + (size_t)(512 + j0) * HID + kb * 32, HID);
      cr = WMMA_F16(a, br_, cr);
      cz = WMMA_F16(a, bz_, cz);
      cn = WMMA_F16(a, bn_, cn);
    }
#pragma unroll
    for (int r = 0; r < 8; ++r) {
      int m = rb + r;
      int bidx = mblk * 16 + m;
      const float* gir = giD + ((size_t)s * BB + bidx) * G3;
      float rg = sigmf(gir[j]       + cr[r] + br);
      float zg = sigmf(gir[256 + j] + cz[r] + bz);
      float ng = tanhf(gir[512 + j] + rg * (cn[r] + bn));
      float hold = (float)h16[cur][m][j];
      float hnew = (1.f - zg) * ng + zg * hold;
      h16[nxt][m][j] = (_Float16)hnew;
      size_t off = ((size_t)bidx * SS + sout) * H2 + (size_t)dir * HID + j;
      hcat[off]   = hnew;
      hcat_h[off] = (_Float16)hnew;
    }
    __syncthreads();
  }
}

// ---------------- K5: attention scores u[b][s] ----------------
// GEMM M=8192, N=ATTP=512, K=512, fused tanh + weighted column reduction.
// grid = 64, block = 256; one wave per 16-row M tile.
__global__ void __launch_bounds__(256) att_score_kernel(
    const _Float16* __restrict__ hcat_h, const _Float16* __restrict__ attW_h,
    const float* __restrict__ att_fc_b, const float* __restrict__ att_weight,
    const int* __restrict__ z, float* __restrict__ u) {
  int lane = threadIdx.x & 31, wave = threadIdx.x >> 5;
  int mtile = blockIdx.x * 8 + wave;            // 0..511
  int m0 = mtile * 16;                          // flat row = b*S + s ; const b per tile
  int b = m0 / SS;
  const _Float16* A = hcat_h + (size_t)m0 * H2;
  const float* aw = att_weight + (size_t)z[b] * ATT;

  float usum[8] = {0.f, 0.f, 0.f, 0.f, 0.f, 0.f, 0.f, 0.f};
  for (int nt = 0; nt < ATTP / 16; ++nt) {
    v8f acc = {};
    for (int kt = 0; kt < H2 / 32; ++kt) {
      v16h a  = load_frag(A + kt * 32, H2);
      v16h bf = load_frag(attW_h + (size_t)(nt * 16) * H2 + kt * 32, H2);
      acc = WMMA_F16(a, bf, acc);
    }
    int n = lane & 15;
    int aidx = nt * 16 + n;
    float w  = (aidx < ATT) ? aw[aidx] : 0.f;
    float bb = (aidx < ATT) ? att_fc_b[aidx] : 0.f;
#pragma unroll
    for (int r = 0; r < 8; ++r) usum[r] += tanhf(acc[r] + bb) * w;
  }
  // reduce over the 16 lanes holding the same row set (xor masks stay in half)
#pragma unroll
  for (int r = 0; r < 8; ++r) {
    float v = usum[r];
    for (int msk = 1; msk < 16; msk <<= 1) v += __shfl_xor(v, msk, 32);
    if ((lane & 15) == 0) u[m0 + (lane >> 4) * 8 + r] = v;
  }
}

// ---------------- K6: softmax over S, attention pooling, final FC ----------------
// grid = 64 (one block per batch row), block = 256.
__global__ void __launch_bounds__(256) finalize_kernel(
    const float* __restrict__ u, const float* __restrict__ hcat,
    const float* __restrict__ fcW, const float* __restrict__ fcb,
    float* __restrict__ out) {
  __shared__ float sred[256];
  __shared__ float satt[SS];
  __shared__ float pooled[H2];
  int b = blockIdx.x;
  int tid = threadIdx.x;

  float v = (tid < SS) ? u[b * SS + tid] : -1e30f;
  sred[tid] = v;
  __syncthreads();
  for (int off = 128; off >= 1; off >>= 1) {
    if (tid < off) sred[tid] = fmaxf(sred[tid], sred[tid + off]);
    __syncthreads();
  }
  float mx = sred[0];
  __syncthreads();
  float e = (tid < SS) ? __expf(v - mx) : 0.f;
  sred[tid] = e;
  __syncthreads();
  for (int off = 128; off >= 1; off >>= 1) {
    if (tid < off) sred[tid] += sred[tid + off];
    __syncthreads();
  }
  float denom = sred[0];
  __syncthreads();
  if (tid < SS) satt[tid] = e / denom;
  __syncthreads();

  const float* hb = hcat + (size_t)b * SS * H2;
  for (int j = tid; j < H2; j += 256) {
    float acc = 0.f;
    for (int s = 0; s < SS; ++s) acc += satt[s] * hb[(size_t)s * H2 + j];
    pooled[j] = acc;
  }
  __syncthreads();

  for (int o = 0; o < 2; ++o) {
    float part = 0.f;
    for (int j = tid; j < H2; j += 256) part += pooled[j] * fcW[o * H2 + j];
    sred[tid] = part;
    __syncthreads();
    for (int off = 128; off >= 1; off >>= 1) {
      if (tid < off) sred[tid] += sred[tid + off];
      __syncthreads();
    }
    if (tid == 0) out[b * 2 + o] = sred[0] + fcb[o];
    __syncthreads();
  }
}

// ---------------- host launch ----------------
extern "C" void kernel_launch(void* const* d_in, const int* in_sizes, int n_in,
                              void* d_out, int out_size, void* d_ws, size_t ws_size,
                              hipStream_t stream) {
  (void)in_sizes; (void)n_in; (void)out_size; (void)ws_size;
  const int*   x          = (const int*)d_in[0];
  const int*   z          = (const int*)d_in[1];
  const float* emb        = (const float*)d_in[2];
  const float* Wih_f      = (const float*)d_in[3];
  const float* Whh_f      = (const float*)d_in[4];
  const float* bih_f      = (const float*)d_in[5];
  const float* bhh_f      = (const float*)d_in[6];
  const float* Wih_b      = (const float*)d_in[7];
  const float* Whh_b      = (const float*)d_in[8];
  const float* bih_b      = (const float*)d_in[9];
  const float* bhh_b      = (const float*)d_in[10];
  const float* att_fc_W   = (const float*)d_in[11];
  const float* att_fc_b   = (const float*)d_in[12];
  const float* att_weight = (const float*)d_in[13];
  const float* fc_W       = (const float*)d_in[14];
  const float* fc_b       = (const float*)d_in[15];
  float* out = (float*)d_out;

  char* wp = (char*)d_ws;
  auto alloc = [&](size_t bytes) -> char* {
    char* p = wp;
    wp += (bytes + 255) & ~(size_t)255;
    return p;
  };
  _Float16* xs_h   = (_Float16*)alloc((size_t)SS * BB * VECP * 2);
  _Float16* Wih_h  = (_Float16*)alloc((size_t)2 * G3 * VECP * 2);
  _Float16* Whh_h  = (_Float16*)alloc((size_t)2 * G3 * HID * 2);
  _Float16* attW_h = (_Float16*)alloc((size_t)ATTP * H2 * 2);
  float*    gi     = (float*)alloc((size_t)2 * SS * BB * G3 * 4);
  float*    hcat   = (float*)alloc((size_t)BB * SS * H2 * 4);
  _Float16* hcat_h = (_Float16*)alloc((size_t)BB * SS * H2 * 2);
  float*    u      = (float*)alloc((size_t)BB * SS * 4);

  // K1: embedding gather
  {
    int total = SS * BB * VECP;
    embed_kernel<<<(total + 255) / 256, 256, 0, stream>>>(x, emb, xs_h);
  }
  // K2: weight conversions
  pad_convert<<<(G3 * VECP + 255) / 256, 256, 0, stream>>>(Wih_f, Wih_h, G3, VEC, G3, VECP);
  pad_convert<<<(G3 * VECP + 255) / 256, 256, 0, stream>>>(Wih_b, Wih_h + (size_t)G3 * VECP, G3, VEC, G3, VECP);
  pad_convert<<<(G3 * HID + 255) / 256, 256, 0, stream>>>(Whh_f, Whh_h, G3, HID, G3, HID);
  pad_convert<<<(G3 * HID + 255) / 256, 256, 0, stream>>>(Whh_b, Whh_h + (size_t)G3 * HID, G3, HID, G3, HID);
  pad_convert<<<(ATTP * H2 + 255) / 256, 256, 0, stream>>>(att_fc_W, attW_h, ATT, H2, ATTP, H2);
  // K3: input projections (both directions), W slab staged via TDM
  igemm_kernel<<<dim3(12, 64, 2), 256, 0, stream>>>(xs_h, Wih_h, bih_f, bih_b, gi);
  // K4: GRU scan (2 dirs x 4 batch tiles), 16 waves per block
  gru_scan_kernel<<<8, 512, 0, stream>>>(gi, Whh_h, bhh_f, bhh_b, hcat, hcat_h);
  // K5: attention scores
  att_score_kernel<<<64, 256, 0, stream>>>(hcat_h, attW_h, att_fc_b, att_weight, z, u);
  // K6: softmax + pooling + FC
  finalize_kernel<<<64, 256, 0, stream>>>(u, hcat, fc_W, fc_b, out);
}
